// transformer_pointer_network_decoder_21955872817718
// MI455X (gfx1250) — matrix operand order repaired and checked by reading, not compiled, not run on hardware
//
#include <hip/hip_runtime.h>
#include <hip/hip_bf16.h>
#include <math.h>

typedef __attribute__((ext_vector_type(16))) __bf16 v16bf;
typedef __attribute__((ext_vector_type(8)))  float  v8f;

#define D_    512
#define H_    8
#define DH_   64
#define DFF_  2048
#define L_    6
#define AA_   64
#define T_    12
#define B_    16
#define MT_   63      // memory tokens (A-1)
#define HID3_ 384

__device__ __forceinline__ void pack8(v16bf& dst, int base, float4 a, float4 b) {
  dst[base + 0] = (__bf16)a.x; dst[base + 1] = (__bf16)a.y;
  dst[base + 2] = (__bf16)a.z; dst[base + 3] = (__bf16)a.w;
  dst[base + 4] = (__bf16)b.x; dst[base + 5] = (__bf16)b.y;
  dst[base + 6] = (__bf16)b.z; dst[base + 7] = (__bf16)b.w;
}

// ---------------------------------------------------------------------------
// GEMM: C[M,N] = act(A[M,K] @ W[N,K]^T + bias[N]) ; f32 in/out, bf16 WMMA.
// One wave -> 16x32 output (one A fragment, two B fragments, 2 WMMAs/chunk).
// Requires: M % 16 == 0, N % 32 == 0, K % 32 == 0 (true for all call sites).
// Fragment layouts used here (wave32, contiguous k order in A and B):
//   A (16x32 bf16): m=lane&15, kb=(lane>>4)*16; elem j -> K = kb + j
//   B (32x16 bf16): n=lane&15, kb=(lane>>4)*16; elem j -> K = kb + j
//   C/D (16x16 f32): vgpr r -> M = r + 8*(lane>>4), N = lane&15
// ---------------------------------------------------------------------------
__global__ __launch_bounds__(256) void gemm_bf16_wmma(
    const float* __restrict__ A, const float* __restrict__ W,
    const float* __restrict__ bias, float* __restrict__ C,
    int M, int K, int N, int relu)
{
  int wave = (int)((blockIdx.x * blockDim.x + threadIdx.x) >> 5);
  int lane = (int)(threadIdx.x & 31);
  int tilesN2 = N >> 5;     // 32-wide tile pairs
  int tilesM  = M >> 4;
  if (wave >= tilesM * tilesN2) return;
  int tm = wave / tilesN2;
  int tn = (wave % tilesN2) * 2;

  int mrow  = tm * 16 + (lane & 15);
  int n0    = tn * 16 + (lane & 15);
  int kAoff = (lane >> 4) * 16;
  int kBoff = (lane >> 4) * 16;

  const float* arow = A + (size_t)mrow * K;
  const float* wr0  = W + (size_t)n0 * K;
  const float* wr1  = wr0 + (size_t)16 * K;

  v8f acc0 = {}, acc1 = {};
  for (int kc = 0; kc < K; kc += 32) {
    v16bf af, bf0, bf1;
    {
      const float4* p0 = (const float4*)(arow + kc + kAoff);
      const float4* p1 = (const float4*)(arow + kc + kAoff + 8);
      pack8(af, 0, p0[0], p0[1]);
      pack8(af, 8, p1[0], p1[1]);
    }
    {
      const float4* p = (const float4*)(wr0 + kc + kBoff);
      pack8(bf0, 0, p[0], p[1]);
      pack8(bf0, 8, p[2], p[3]);
    }
    {
      const float4* p = (const float4*)(wr1 + kc + kBoff);
      pack8(bf1, 0, p[0], p[1]);
      pack8(bf1, 8, p[2], p[3]);
    }
    acc0 = __builtin_amdgcn_wmma_f32_16x16x32_bf16(
        false, af, false, bf0, (short)0, acc0, false, false);
    acc1 = __builtin_amdgcn_wmma_f32_16x16x32_bf16(
        false, af, false, bf1, (short)0, acc1, false, false);
  }

  float bv0 = bias[n0];
  float bv1 = bias[n0 + 16];
  int rbase = tm * 16 + ((lane >> 4) * 8);
#pragma unroll
  for (int r = 0; r < 8; ++r) {
    float v0 = acc0[r] + bv0;
    float v1 = acc1[r] + bv1;
    if (relu) { v0 = fmaxf(v0, 0.0f); v1 = fmaxf(v1, 0.0f); }
    C[(size_t)(rbase + r) * N + n0]      = v0;
    C[(size_t)(rbase + r) * N + n0 + 16] = v1;
  }
}

// ---------------------------------------------------------------------------
// Online-softmax attention: one thread per (b, h, q). dh = 64, scale = 1/8.
// Q rows: b*nq + t ; K/V rows: b*nk + j (each row D_ floats, head at h*DH_).
// ---------------------------------------------------------------------------
__global__ __launch_bounds__(64) void attn_kernel(
    const float* __restrict__ Q, int nq,
    const float* __restrict__ Km, const float* __restrict__ Vm, int nk,
    float* __restrict__ O)
{
  int tid = (int)(blockIdx.x * blockDim.x + threadIdx.x);
  int total = B_ * H_ * nq;
  if (tid >= total) return;
  int t = tid % nq;
  int h = (tid / nq) % H_;
  int b = tid / (nq * H_);

  const float4* qp = (const float4*)(Q + (size_t)(b * nq + t) * D_ + h * DH_);
  float4 qr[16];
#pragma unroll
  for (int e = 0; e < 16; ++e) qr[e] = qp[e];

  float mmax = -INFINITY, ssum = 0.0f;
  float4 oacc[16];
#pragma unroll
  for (int e = 0; e < 16; ++e) oacc[e] = make_float4(0.f, 0.f, 0.f, 0.f);

  for (int j = 0; j < nk; ++j) {
    const float4* kp = (const float4*)(Km + (size_t)(b * nk + j) * D_ + h * DH_);
    float s = 0.0f;
#pragma unroll
    for (int e = 0; e < 16; ++e) {
      float4 kv = kp[e];
      s += qr[e].x * kv.x + qr[e].y * kv.y + qr[e].z * kv.z + qr[e].w * kv.w;
    }
    s *= 0.125f;  // 1/sqrt(64)
    float nm   = fmaxf(mmax, s);
    float corr = __expf(mmax - nm);   // 0 on first iter (mmax = -inf)
    float w    = __expf(s - nm);
    const float4* vp = (const float4*)(Vm + (size_t)(b * nk + j) * D_ + h * DH_);
#pragma unroll
    for (int e = 0; e < 16; ++e) {
      float4 vv = vp[e];
      oacc[e].x = oacc[e].x * corr + w * vv.x;
      oacc[e].y = oacc[e].y * corr + w * vv.y;
      oacc[e].z = oacc[e].z * corr + w * vv.z;
      oacc[e].w = oacc[e].w * corr + w * vv.w;
    }
    ssum = ssum * corr + w;
    mmax = nm;
  }
  float inv = 1.0f / ssum;
  float4* op = (float4*)(O + (size_t)(b * nq + t) * D_ + h * DH_);
#pragma unroll
  for (int e = 0; e < 16; ++e) {
    float4 o4 = oacc[e];
    op[e] = make_float4(o4.x * inv, o4.y * inv, o4.z * inv, o4.w * inv);
  }
}

// ---------------------------------------------------------------------------
// LayerNorm over D_=512; one wave per row; lane owns 16 contiguous floats.
// ---------------------------------------------------------------------------
__device__ __forceinline__ void ln_row_body(
    const float* in, const float* res, float* out,
    const float* g, const float* bta, int lane)
{
  const int base = lane * 16;
  float4 v[4];
  float s = 0.0f;
#pragma unroll
  for (int j = 0; j < 4; ++j) {
    float4 t = ((const float4*)(in + base))[j];
    if (res) {
      float4 rr = ((const float4*)(res + base))[j];
      t.x += rr.x; t.y += rr.y; t.z += rr.z; t.w += rr.w;
    }
    v[j] = t;
    s += t.x + t.y + t.z + t.w;
  }
#pragma unroll
  for (int off = 16; off >= 1; off >>= 1) s += __shfl_xor(s, off, 32);
  float mean = s * (1.0f / 512.0f);
  float vs = 0.0f;
#pragma unroll
  for (int j = 0; j < 4; ++j) {
    float dx = v[j].x - mean, dy = v[j].y - mean, dz = v[j].z - mean, dw = v[j].w - mean;
    vs += dx * dx + dy * dy + dz * dz + dw * dw;
  }
#pragma unroll
  for (int off = 16; off >= 1; off >>= 1) vs += __shfl_xor(vs, off, 32);
  float rstd = rsqrtf(vs * (1.0f / 512.0f) + 1e-5f);
#pragma unroll
  for (int j = 0; j < 4; ++j) {
    float4 gg = ((const float4*)(g + base))[j];
    float4 bb = ((const float4*)(bta + base))[j];
    float4 o;
    o.x = (v[j].x - mean) * rstd * gg.x + bb.x;
    o.y = (v[j].y - mean) * rstd * gg.y + bb.y;
    o.z = (v[j].z - mean) * rstd * gg.z + bb.z;
    o.w = (v[j].w - mean) * rstd * gg.w + bb.w;
    ((float4*)(out + base))[j] = o;
  }
}

__global__ __launch_bounds__(256) void add_ln_kernel(
    float* __restrict__ x, const float* __restrict__ res,
    const float* __restrict__ g, const float* __restrict__ bta, int rows)
{
  int wave = (int)((blockIdx.x * blockDim.x + threadIdx.x) >> 5);
  int lane = (int)(threadIdx.x & 31);
  if (wave >= rows) return;
  float* xr = x + (size_t)wave * D_;
  ln_row_body(xr, res + (size_t)wave * D_, xr, g, bta, lane);
}

__global__ __launch_bounds__(256) void ln_last_kernel(
    const float* __restrict__ x, float* __restrict__ out,
    const float* __restrict__ g, const float* __restrict__ bta, int n)
{
  int wave = (int)((blockIdx.x * blockDim.x + threadIdx.x) >> 5);
  int lane = (int)(threadIdx.x & 31);
  if (wave >= B_) return;
  ln_row_body(x + (size_t)(wave * n + n - 1) * D_, nullptr,
              out + (size_t)wave * D_, g, bta, lane);
}

// ---------------------------------------------------------------------------
// Small glue kernels (float4 granularity).
// ---------------------------------------------------------------------------
__global__ void build_concat_kernel(const float4* __restrict__ embtmp,
                                    const float4* __restrict__ init_dec,
                                    float4* __restrict__ concat,
                                    float4* __restrict__ dec_in)
{
  int i = (int)(blockIdx.x * blockDim.x + threadIdx.x);
  const int total = B_ * AA_ * (D_ / 4);
  if (i >= total) return;
  int d = i & (D_ / 4 - 1);
  int r = i >> 7;
  int s = r & (AA_ - 1);
  int b = r >> 6;
  float4 val = (s == 0) ? init_dec[d]
                        : embtmp[(size_t)(b * MT_ + s - 1) * (D_ / 4) + d];
  concat[i] = val;
  if (s == 0) dec_in[(size_t)(b * (T_ + 1)) * (D_ / 4) + d] = val;  // dec_in[b,0,:]
}

__global__ void gather_x_kernel(const float4* __restrict__ dec_in,
                                float4* __restrict__ x, int n)
{
  int i = (int)(blockIdx.x * blockDim.x + threadIdx.x);
  int total = B_ * n * (D_ / 4);
  if (i >= total) return;
  int d = i & (D_ / 4 - 1);
  int r = i >> 7;
  int t = r % n;
  int b = r / n;
  x[i] = dec_in[(size_t)(b * (T_ + 1) + t) * (D_ / 4) + d];
}

__global__ void init_kernel(int* __restrict__ mask, int* __restrict__ term,
                            float* __restrict__ lj)
{
  int i = (int)(blockIdx.x * blockDim.x + threadIdx.x);
  if (i < B_ * AA_) mask[i] = 0;
  if (i < B_) { term[i] = 0; lj[i] = 0.0f; }
}

// Pointer softmax + bookkeeping + token append. One block per batch element.
__global__ __launch_bounds__(64) void step_update_kernel(
    const float* __restrict__ logits, const int* __restrict__ actions, int step,
    int* __restrict__ mask, int* __restrict__ term, float* __restrict__ lj,
    const float* __restrict__ concat, float* __restrict__ dec_in)
{
  int b = (int)blockIdx.x;
  int a = (int)threadIdx.x;
  __shared__ float sh[AA_];
  __shared__ int idx_sh;

  float x = logits[b * AA_ + a];
  if (mask[b * AA_ + a]) x = -INFINITY;

  sh[a] = x; __syncthreads();
  for (int off = 32; off; off >>= 1) {
    if (a < off) sh[a] = fmaxf(sh[a], sh[a + off]);
    __syncthreads();
  }
  float mx = sh[0]; __syncthreads();

  float e = __expf(x - mx);
  sh[a] = e; __syncthreads();
  for (int off = 32; off; off >>= 1) {
    if (a < off) sh[a] += sh[a + off];
    __syncthreads();
  }
  float sum = sh[0]; __syncthreads();

  sh[a] = e / sum; __syncthreads();

  if (a == 0) {
    int idx = actions[b * T_ + step];
    float prob = sh[idx];
    float lp = term[b] ? 0.0f : __logf(prob + 1e-7f);
    lj[b] += lp;
    mask[b * AA_ + idx] = 1;        // mask updated AFTER prob taken
    term[b] = (idx == 0) ? 1 : 0;   // new term for next step
    idx_sh = idx;
  }
  __syncthreads();
  int idx = idx_sh;
  const float4* src = (const float4*)(concat + (size_t)(b * AA_ + idx) * D_);
  float4* dst = (float4*)(dec_in + (size_t)(b * (T_ + 1) + step + 1) * D_);
  for (int d = a; d < D_ / 4; d += 64) dst[d] = src[d];
}

// ---------------------------------------------------------------------------
// Host orchestration
// ---------------------------------------------------------------------------
extern "C" void kernel_launch(void* const* d_in, const int* in_sizes, int n_in,
                              void* d_out, int out_size, void* d_ws, size_t ws_size,
                              hipStream_t stream)
{
  (void)in_sizes; (void)n_in; (void)out_size; (void)ws_size;

  const float* tgt      = (const float*)d_in[0];   // (B, 63, 384)
  const float* memory   = (const float*)d_in[1];   // (B, 63, 512)
  const float* emb_w    = (const float*)d_in[2];   // (512, 384)
  const float* emb_b    = (const float*)d_in[3];
  const float* init_dec = (const float*)d_in[4];
  const float* aff_w    = (const float*)d_in[5];   // (64, 512)
  const float* aff_b    = (const float*)d_in[6];
  const float* Wq_s = (const float*)d_in[7];
  const float* Wk_s = (const float*)d_in[8];
  const float* Wv_s = (const float*)d_in[9];
  const float* Wo_s = (const float*)d_in[10];
  const float* Wq_c = (const float*)d_in[11];
  const float* Wk_c = (const float*)d_in[12];
  const float* Wv_c = (const float*)d_in[13];
  const float* Wo_c = (const float*)d_in[14];
  const float* bq_s = (const float*)d_in[15];
  const float* bk_s = (const float*)d_in[16];
  const float* bv_s = (const float*)d_in[17];
  const float* bo_s = (const float*)d_in[18];
  const float* bq_c = (const float*)d_in[19];
  const float* bk_c = (const float*)d_in[20];
  const float* bv_c = (const float*)d_in[21];
  const float* bo_c = (const float*)d_in[22];
  const float* w1   = (const float*)d_in[23];      // (L, 2048, 512)
  const float* b1   = (const float*)d_in[24];
  const float* w2   = (const float*)d_in[25];      // (L, 512, 2048)
  const float* b2   = (const float*)d_in[26];
  const float* ln_g = (const float*)d_in[27];      // (L, 3, 512)
  const float* ln_b = (const float*)d_in[28];
  const float* gf   = (const float*)d_in[29];
  const float* bf   = (const float*)d_in[30];
  const int*   actions = (const int*)d_in[31];     // (B, T, 1)

  float* lj = (float*)d_out;                       // (B,) accumulated log-joint

  // Workspace carve-up (floats)
  float* wsf = (float*)d_ws;
  size_t off = 0;
  auto alloc = [&](size_t nf) { float* p = wsf + off; off += nf; return p; };
  const int MR = B_ * MT_;                         // 1008 memory rows
  float* embtmp = alloc((size_t)MR * D_);
  float* concat = alloc((size_t)B_ * AA_ * D_);
  float* crossK = alloc((size_t)L_ * MR * D_);
  float* crossV = alloc((size_t)L_ * MR * D_);
  float* dec_in = alloc((size_t)B_ * (T_ + 1) * D_);
  float* xbuf   = alloc((size_t)B_ * T_ * D_);
  float* qb     = alloc((size_t)B_ * T_ * D_);
  float* kbuf   = alloc((size_t)B_ * T_ * D_);
  float* vbuf   = alloc((size_t)B_ * T_ * D_);
  float* obuf   = alloc((size_t)B_ * T_ * D_);
  float* tmpb   = alloc((size_t)B_ * T_ * D_);
  float* h1b    = alloc((size_t)B_ * T_ * DFF_);
  float* hlast  = alloc((size_t)B_ * D_);
  float* logits = alloc((size_t)B_ * AA_);
  int* mask = (int*)(wsf + off); off += B_ * AA_;
  int* term = (int*)(wsf + off); off += B_;

  const size_t DD  = (size_t)D_ * D_;
  const size_t WFF = (size_t)DFF_ * D_;

  auto gemm = [&](const float* Ap, const float* Wp, const float* bp, float* Cp,
                  int M, int K, int N, int relu) {
    int tiles = (M / 16) * (N / 32);
    int blocks = (tiles + 7) / 8;
    gemm_bf16_wmma<<<dim3(blocks), dim3(256), 0, stream>>>(Ap, Wp, bp, Cp, M, K, N, relu);
  };

  // ---- setup phase ----
  init_kernel<<<dim3(4), dim3(256), 0, stream>>>(mask, term, lj);

  // emb = relu(tgt @ emb_w^T + emb_b) : (1008 x 384) @ (384 x 512)
  gemm(tgt, emb_w, emb_b, embtmp, MR, HID3_, D_, 1);
  build_concat_kernel<<<dim3((B_ * AA_ * (D_ / 4) + 255) / 256), dim3(256), 0, stream>>>(
      (const float4*)embtmp, (const float4*)init_dec, (float4*)concat, (float4*)dec_in);

  // precompute cross-attention K/V per layer (memory is step-invariant)
  for (int l = 0; l < L_; ++l) {
    gemm(memory, Wk_c + l * DD, bk_c + l * D_, crossK + (size_t)l * MR * D_, MR, D_, D_, 0);
    gemm(memory, Wv_c + l * DD, bv_c + l * D_, crossV + (size_t)l * MR * D_, MR, D_, D_, 0);
  }

  // ---- sequential decode ----
  for (int i = 0; i < T_; ++i) {
    int n = i + 1;
    int R = B_ * n;

    gather_x_kernel<<<dim3((R * (D_ / 4) + 255) / 256), dim3(256), 0, stream>>>(
        (const float4*)dec_in, (float4*)xbuf, n);

    for (int l = 0; l < L_; ++l) {
      // self-attention
      gemm(xbuf, Wq_s + l * DD, bq_s + l * D_, qb,   R, D_, D_, 0);
      gemm(xbuf, Wk_s + l * DD, bk_s + l * D_, kbuf, R, D_, D_, 0);
      gemm(xbuf, Wv_s + l * DD, bv_s + l * D_, vbuf, R, D_, D_, 0);
      attn_kernel<<<dim3((B_ * H_ * n + 63) / 64), dim3(64), 0, stream>>>(
          qb, n, kbuf, vbuf, n, obuf);
      gemm(obuf, Wo_s + l * DD, bo_s + l * D_, tmpb, R, D_, D_, 0);
      add_ln_kernel<<<dim3((R + 7) / 8), dim3(256), 0, stream>>>(
          xbuf, tmpb, ln_g + (size_t)(l * 3 + 0) * D_, ln_b + (size_t)(l * 3 + 0) * D_, R);

      // cross-attention (K/V precomputed)
      gemm(xbuf, Wq_c + l * DD, bq_c + l * D_, qb, R, D_, D_, 0);
      attn_kernel<<<dim3((B_ * H_ * n + 63) / 64), dim3(64), 0, stream>>>(
          qb, n, crossK + (size_t)l * MR * D_, crossV + (size_t)l * MR * D_, MT_, obuf);
      gemm(obuf, Wo_c + l * DD, bo_c + l * D_, tmpb, R, D_, D_, 0);
      add_ln_kernel<<<dim3((R + 7) / 8), dim3(256), 0, stream>>>(
          xbuf, tmpb, ln_g + (size_t)(l * 3 + 1) * D_, ln_b + (size_t)(l * 3 + 1) * D_, R);

      // FFN
      gemm(xbuf, w1 + l * WFF, b1 + (size_t)l * DFF_, h1b, R, D_, DFF_, 1);
      gemm(h1b, w2 + l * WFF, b2 + (size_t)l * D_, tmpb, R, DFF_, D_, 0);
      add_ln_kernel<<<dim3((R + 7) / 8), dim3(256), 0, stream>>>(
          xbuf, tmpb, ln_g + (size_t)(l * 3 + 2) * D_, ln_b + (size_t)(l * 3 + 2) * D_, R);
    }

    // final LN of last token, pointer logits, bookkeeping + append
    ln_last_kernel<<<dim3(2), dim3(256), 0, stream>>>(xbuf, hlast, gf, bf, n);
    gemm(hlast, aff_w, aff_b, logits, B_, D_, AA_, 0);
    step_update_kernel<<<dim3(B_), dim3(64), 0, stream>>>(
        logits, actions, i, mask, term, lj, concat, dec_in);
  }
}
